// SampledSegmentDictDMPV1_55379308315485
// MI455X (gfx1250) — compile-verified, matrix operand order checked by reading, not run on hardware
//
#include <hip/hip_runtime.h>
#include <math.h>

typedef __attribute__((ext_vector_type(16))) _Float16 v16h;
typedef __attribute__((ext_vector_type(8)))  _Float16 v8h;
typedef __attribute__((ext_vector_type(8)))  float    v8f;

#define TPB 256

// ---------------- elementwise / conversion kernels ----------------

__global__ void k_cast_f32_f16(const float* __restrict__ in, _Float16* __restrict__ out,
                               long long n) {
  long long i = (long long)blockIdx.x * blockDim.x + threadIdx.x;
  if (i < n) out[i] = (_Float16)in[i];
}

// weight (N x K) f32 row-major -> (N x Kpad) f16, zero-padded along K
__global__ void k_cvt_weight(const float* __restrict__ w, _Float16* __restrict__ o,
                             int N, int K, int Kpad) {
  long long i = (long long)blockIdx.x * blockDim.x + threadIdx.x;
  long long tot = (long long)N * Kpad;
  if (i >= tot) return;
  int k = (int)(i % Kpad);
  int n = (int)(i / Kpad);
  o[i] = (k < K) ? (_Float16)w[(long long)n * K + k] : (_Float16)0.0f;
}

// NCHW f32 -> im2col rows (M = B*Hc*Wc, K = Cin*9 padded to Kpad), f16
__global__ void k_im2col(const float* __restrict__ in, _Float16* __restrict__ out,
                         int Bn, int Cin, int Hin, int Win, int Hc, int Wc,
                         int K, int Kpad) {
  long long i = (long long)blockIdx.x * blockDim.x + threadIdx.x;
  long long tot = (long long)Bn * Hc * Wc * Kpad;
  if (i >= tot) return;
  int k = (int)(i % Kpad);
  long long m = i / Kpad;
  int x = (int)(m % Wc);
  int y = (int)((m / Wc) % Hc);
  int b = (int)(m / ((long long)Wc * Hc));
  if (k >= K) { out[i] = (_Float16)0.0f; return; }
  int ci = k / 9, r = k % 9, ky = r / 3, kx = r % 3;
  out[i] = (_Float16)in[(((long long)b * Cin + ci) * Hin + (y + ky)) * Win + (x + kx)];
}

// ---------------- WMMA GEMM: C(MxN,f32) = A(MxK,f16,row-major) * B^T (B stored N x K, f16)
// Per wave: one 16x64 tile. k-loop unrolled x2 with two independent fragment sets
// (ping-pong by loop structure, no register copies): while WMMAs consume set0, the
// prefetch loads for set0's next use are in flight behind the set1 WMMAs and vice
// versa, so the waitcnt pass can issue partial loadcnt waits. K must be % 64 == 0.
// Prefetch is unconditional: the final iteration overreads <=128B into the arena.
struct FragAB { v8h lo, hi; };

__device__ __forceinline__ FragAB ldfrag(const _Float16* __restrict__ p, unsigned off) {
  FragAB f;
  f.lo = *(const v8h*)(p + off);
  f.hi = *(const v8h*)(p + off + 16);
  return f;
}

__device__ __forceinline__ v16h catfrag(FragAB f) {
  v16h a;
#pragma unroll
  for (int t = 0; t < 8; ++t) { a[t] = f.lo[t]; a[t + 8] = f.hi[t]; }
  return a;
}

__global__ __launch_bounds__(256)
void k_gemm_wmma(const _Float16* __restrict__ A, const _Float16* __restrict__ Bm,
                 float* __restrict__ C, int M, int N, int K) {
  const int lane = threadIdx.x & 31;
  const int wave = threadIdx.x >> 5;
  const int tilesN = N >> 6;                 // N/64
  const int tile = blockIdx.x * 8 + wave;
  const int totalTiles = (M >> 4) * tilesN;
  if (tile >= totalTiles) return;            // wave-uniform exit: EXEC stays all-ones
  const int tm = tile / tilesN;
  const int tn = tile % tilesN;
  const int row = lane & 15;
  const int hi  = (lane >> 4) << 3;          // 0 or 8 (halves)

  const unsigned uK   = (unsigned)K;
  const unsigned aoff = (unsigned)(tm * 16 + row) * uK + (unsigned)hi;
  const unsigned boff = (unsigned)(tn * 64 + row) * uK + (unsigned)hi;

  FragAB a0 = ldfrag(A, aoff);
  FragAB b0[4];
#pragma unroll
  for (int j = 0; j < 4; ++j) b0[j] = ldfrag(Bm, boff + (unsigned)j * 16u * uK);
  FragAB a1 = ldfrag(A, aoff + 32);
  FragAB b1[4];
#pragma unroll
  for (int j = 0; j < 4; ++j) b1[j] = ldfrag(Bm, boff + (unsigned)j * 16u * uK + 32);

  v8f acc[4] = {};
  for (int k = 0; k < K; k += 64) {
    {
      v16h av = catfrag(a0);
#pragma unroll
      for (int j = 0; j < 4; ++j)
        acc[j] = __builtin_amdgcn_wmma_f32_16x16x32_f16(false, av, false, catfrag(b0[j]),
                                                        (short)0, acc[j], false, false);
    }
    a0 = ldfrag(A, aoff + (unsigned)k + 64);
#pragma unroll
    for (int j = 0; j < 4; ++j)
      b0[j] = ldfrag(Bm, boff + (unsigned)j * 16u * uK + (unsigned)k + 64);
    {
      v16h av = catfrag(a1);
#pragma unroll
      for (int j = 0; j < 4; ++j)
        acc[j] = __builtin_amdgcn_wmma_f32_16x16x32_f16(false, av, false, catfrag(b1[j]),
                                                        (short)0, acc[j], false, false);
    }
    a1 = ldfrag(A, aoff + (unsigned)k + 96);
#pragma unroll
    for (int j = 0; j < 4; ++j)
      b1[j] = ldfrag(Bm, boff + (unsigned)j * 16u * uK + (unsigned)k + 96);
  }

  // C/D layout: VGPR r, lanes 0-15 -> M=r, N=lane; lanes 16-31 -> M=8+r, N=lane-16
  const int cm  = tm * 16 + hi;              // hi==8 -> rows 8..15
  const int cn0 = tn * 64 + (lane & 15);
#pragma unroll
  for (int j = 0; j < 4; ++j) {
    float* Cp = C + (long long)cm * N + cn0 + j * 16;
#pragma unroll
    for (int r = 0; r < 8; ++r) Cp[(long long)r * N] = acc[j][r];
  }
}

// ---------------- bias + 2x2 maxpool + relu; gemm out (M=B*Hc*Wc, N=Co) -> NCHW pooled ----
__global__ void k_bias_pool_relu(const float* __restrict__ g, const float* __restrict__ bias,
                                 float* __restrict__ out, int Bn, int Co, int Hc, int Wc,
                                 int Hp, int Wp) {
  long long i = (long long)blockIdx.x * blockDim.x + threadIdx.x;
  long long tot = (long long)Bn * Co * Hp * Wp;
  if (i >= tot) return;
  int px = (int)(i % Wp);
  int py = (int)((i / Wp) % Hp);
  int c  = (int)((i / ((long long)Wp * Hp)) % Co);
  int b  = (int)(i / ((long long)Wp * Hp * Co));
  float m = -3.402823466e38f;
  for (int dy = 0; dy < 2; ++dy)
    for (int dx = 0; dx < 2; ++dx) {
      long long mm = ((long long)b * Hc + (2 * py + dy)) * Wc + (2 * px + dx);
      m = fmaxf(m, g[mm * Co + c]);
    }
  out[i] = fmaxf(m + bias[c], 0.0f);
}

__global__ void k_tanh_bias(const float* __restrict__ g, const float* __restrict__ bias,
                            float* __restrict__ z, _Float16* __restrict__ zh,
                            int n, int hid) {
  int i = blockIdx.x * blockDim.x + threadIdx.x;
  if (i >= n) return;
  float v = tanhf(g[i] + bias[i % hid]);
  z[i] = v; zh[i] = (_Float16)v;
}

__global__ void k_add_bias(const float* __restrict__ g, const float* __restrict__ bias,
                           float* __restrict__ z, int n, int hid) {
  int i = blockIdx.x * blockDim.x + threadIdx.x;
  if (i >= n) return;
  z[i] = g[i] + bias[i % hid];
}

// ---------------- heads: init_pos, seg_mod, ratio(softmax). one thread per batch ----------
__global__ void k_heads(const float* __restrict__ z,
                        const float* __restrict__ wi, const float* __restrict__ bi,
                        const float* __restrict__ wseg, const float* __restrict__ bseg,
                        const float* __restrict__ wrr, const float* __restrict__ brr,
                        float* __restrict__ initPos, float* __restrict__ segMod,
                        float* __restrict__ ratioWs, float* __restrict__ ratioOut) {
  int b = threadIdx.x;
  if (b >= 32) return;
  const float* zb = z + b * 2048;
  for (int o = 0; o < 2; ++o) {
    float s = bi[o];
    for (int k = 0; k < 2048; ++k) s += zb[k] * wi[o * 2048 + k];
    initPos[b * 2 + o] = s;
  }
  for (int o = 0; o < 30; ++o) {
    float s = bseg[o];
    for (int k = 0; k < 2048; ++k) s += zb[k] * wseg[o * 2048 + k];
    segMod[b * 30 + o] = s;
  }
  float r[10]; float mx = -3.402823466e38f;
  for (int o = 0; o < 10; ++o) {
    float s = brr[o];
    for (int k = 0; k < 2048; ++k) s += zb[k] * wrr[o * 2048 + k];
    r[o] = s; mx = fmaxf(mx, s);
  }
  float sum = 0.f;
  for (int o = 0; o < 10; ++o) { r[o] = expf(r[o] - mx); sum += r[o]; }
  for (int o = 0; o < 10; ++o) {
    float v = r[o] / sum;
    ratioWs[b * 10 + o] = v;
    ratioOut[b * 10 + o] = v;
  }
}

// ---------------- trajectory composition; one block per batch ----------------
__global__ void k_traj(const float* __restrict__ segMod, const float* __restrict__ initPos,
                       const float* __restrict__ base, float* __restrict__ trajOut,
                       float* __restrict__ sseOut) {
  int b = blockIdx.x;
  __shared__ float sx[10], sy[10], cs[10], sn[10], stx[11], sty[11];
  if (threadIdx.x == 0) {
    stx[0] = initPos[b * 2 + 0];
    sty[0] = initPos[b * 2 + 1];
    for (int i = 0; i < 10; ++i) {
      sx[i] = segMod[b * 30 + i * 3 + 0];
      sy[i] = segMod[b * 30 + i * 3 + 1];
      float rot = tanhf(segMod[b * 30 + i * 3 + 2]) * 3.14f;
      cs[i] = cosf(rot); sn[i] = sinf(rot);
      float cx = base[0] * sx[i], cy = base[1] * sy[i];            // rotation center = m[:,0,:]
      float ex = base[199 * 2] * sx[i], ey = base[199 * 2 + 1] * sy[i];
      float rx = cs[i] * (ex - cx) - sn[i] * (ey - cy) + cx;
      float ry = sn[i] * (ex - cx) + cs[i] * (ey - cy) + cy;
      stx[i + 1] = stx[i] + rx;
      sty[i + 1] = sty[i] + ry;
    }
  }
  __syncthreads();
  for (int idx = threadIdx.x; idx < 2000; idx += blockDim.x) {
    int i = idx / 200, j = idx % 200;
    float cx = base[0] * sx[i], cy = base[1] * sy[i];
    float mxv = base[j * 2] * sx[i], myv = base[j * 2 + 1] * sy[i];
    float px = cs[i] * (mxv - cx) - sn[i] * (myv - cy) + cx + stx[i];
    float py = sn[i] * (mxv - cx) + cs[i] * (myv - cy) + cy + sty[i];
    trajOut[((long long)b * 2000 + idx) * 2 + 0] = px;
    trajOut[((long long)b * 2000 + idx) * 2 + 1] = py;
  }
  if (threadIdx.x < 11) {
    sseOut[(b * 11 + threadIdx.x) * 2 + 0] = stx[threadIdx.x];
    sseOut[(b * 11 + threadIdx.x) * 2 + 1] = sty[threadIdx.x];
  }
}

// ---------------- sampling (seg-limit trim, 2000-step scan, stable sel-first gather) ------
__global__ void k_sample(const float* __restrict__ ratio, const float* __restrict__ traj,
                         int* __restrict__ mask, float* __restrict__ sampled) {
  int b = threadIdx.x;
  if (b >= 32) return;
  float lim[10];
  for (int j = 0; j < 10; ++j) lim[j] = ceilf(ratio[b * 10 + j] * 200.0f);
  for (int it = 0; it < 10; ++it) {
    float mx = lim[0], sum = lim[0];
    for (int j = 1; j < 10; ++j) { mx = fmaxf(mx, lim[j]); sum += lim[j]; }
    float still = (sum != 200.0f) ? 1.0f : 0.0f;
    float reduced = 1.0f;
    for (int j = 0; j < 10; ++j) {
      float d = mx - lim[j];
      float sg = (d > 0.f) ? 1.f : ((d < 0.f) ? -1.f : 0.f);
      float mult = fabsf(sg - 1.0f);
      lim[j] -= mult * reduced * still;
      reduced = fmaxf(reduced - mult, 0.0f);
    }
  }
  float di = floorf(199.0f / lim[0]);
  float last_idx = 0.f, n_tot = 0.f;
  float cntArr[10];
  for (int j = 0; j < 10; ++j) cntArr[j] = 0.f;
  int* mb = mask + b * 2000;
  for (int i = 0; i < 2000; ++i) {
    int seg = i / 200;
    int fl = (int)floorf((float)i - 0.005f);   // replicates floor(i - 1.0/L)
    bool force = (i > 0) && (seg != fl);
    float L_ = lim[seg], cnt = cntArr[seg];
    float new_di = floorf((199.0f - (last_idx - (float)seg * 200.0f)) / (L_ - cnt));
    if (force || (di < new_di)) di = new_di;
    float c1 = ((200.0f - n_tot) > 0.0f) ? 1.0f : 0.0f;
    float c2 = (cnt < L_) ? 1.0f : 0.0f;
    float c3 = (((float)i - last_idx) == di) ? 1.0f : 0.0f;
    float chk = c1 * c2 * c3;
    last_idx = fmaxf(last_idx, (float)i * chk);
    cntArr[seg] = cnt + chk;
    n_tot += chk;
    mb[i] = (chk > 0.f) ? 1 : 0;
  }
  int outp = 0;
  for (int i = 0; i < 2000 && outp < 200; ++i)
    if (mb[i]) {
      sampled[((long long)b * 200 + outp) * 2 + 0] = traj[((long long)b * 2000 + i) * 2 + 0];
      sampled[((long long)b * 200 + outp) * 2 + 1] = traj[((long long)b * 2000 + i) * 2 + 1];
      ++outp;
    }
  for (int i = 0; i < 2000 && outp < 200; ++i)
    if (!mb[i]) {
      sampled[((long long)b * 200 + outp) * 2 + 0] = traj[((long long)b * 2000 + i) * 2 + 0];
      sampled[((long long)b * 200 + outp) * 2 + 1] = traj[((long long)b * 2000 + i) * 2 + 1];
      ++outp;
    }
}

// ---------------- host orchestration ----------------

extern "C" void kernel_launch(void* const* d_in, const int* in_sizes, int n_in,
                              void* d_out, int out_size, void* d_ws, size_t ws_size,
                              hipStream_t stream) {
  (void)in_sizes; (void)n_in; (void)out_size; (void)ws_size;
  const float* x    = (const float*)d_in[0];
  const float* w1   = (const float*)d_in[1];
  const float* b1   = (const float*)d_in[2];
  const float* w2   = (const float*)d_in[3];
  const float* b2   = (const float*)d_in[4];
  const float* w3   = (const float*)d_in[5];
  const float* b3   = (const float*)d_in[6];
  const float* w4   = (const float*)d_in[7];
  const float* b4   = (const float*)d_in[8];
  const float* fcw1 = (const float*)d_in[9];
  const float* fcb1 = (const float*)d_in[10];
  const float* fcw2 = (const float*)d_in[11];
  const float* fcb2 = (const float*)d_in[12];
  const float* wi   = (const float*)d_in[13];
  const float* bi   = (const float*)d_in[14];
  const float* wseg = (const float*)d_in[15];
  const float* bs   = (const float*)d_in[16];
  const float* wrr  = (const float*)d_in[17];
  const float* brr  = (const float*)d_in[18];
  const float* base = (const float*)d_in[19];

  float* outF       = (float*)d_out;
  float* outSampled = outF;            // 32*200*2 = 12800
  float* outSSE     = outF + 12800;    // 32*11*2  = 704
  float* outRatio   = outF + 13504;    // 32*10    = 320
  float* outTraj    = outF + 13824;    // 32*2000*2 = 128000

  char* wsB = (char*)d_ws;
  size_t off = 0;
  auto alloc = [&](size_t bytes) -> void* {
    void* p = wsB + off;
    off = (off + bytes + 255) & ~(size_t)255;
    return p;
  };

  _Float16* w1h   = (_Float16*)alloc((size_t)64 * 64 * 2);        // Kpad=64
  _Float16* w2h   = (_Float16*)alloc((size_t)128 * 576 * 2);
  _Float16* w3h   = (_Float16*)alloc((size_t)256 * 1152 * 2);
  _Float16* w4h   = (_Float16*)alloc((size_t)512 * 2304 * 2);
  _Float16* f1h   = (_Float16*)alloc((size_t)2048 * 18432 * 2);
  _Float16* f2h   = (_Float16*)alloc((size_t)2048 * 2048 * 2);
  _Float16* imb   = (_Float16*)alloc((size_t)119072 * 576 * 2);   // max im2col (layer2)
  float*    gb    = (float*)alloc((size_t)508032 * 64 * 4);       // max gemm out (layer1)
  float*    poolA = (float*)alloc((size_t)32 * 64 * 63 * 63 * 4);
  float*    poolB = (float*)alloc((size_t)32 * 128 * 30 * 30 * 4);
  _Float16* feath = (_Float16*)alloc((size_t)32 * 18432 * 2);
  float*    z1    = (float*)alloc((size_t)32 * 2048 * 4);
  _Float16* z1h   = (_Float16*)alloc((size_t)32 * 2048 * 2);
  float*    z2    = (float*)alloc((size_t)32 * 2048 * 4);
  float*    initP = (float*)alloc(32 * 2 * 4);
  float*    segM  = (float*)alloc(32 * 30 * 4);
  float*    ratW  = (float*)alloc(32 * 10 * 4);
  int*      maskB = (int*)alloc((size_t)32 * 2000 * 4 + 256);     // +tail slack for overread

  auto nb = [](long long n) { return (unsigned)((n + TPB - 1) / TPB); };
  auto gemmBlocks = [](int M, int N) { return (unsigned)(((M / 16) * (N / 64) + 7) / 8); };

  // weight conversions (N x K f32 -> N x Kpad f16)
  k_cvt_weight<<<nb(64LL * 64), TPB, 0, stream>>>(w1, w1h, 64, 27, 64);
  k_cvt_weight<<<nb(128LL * 576), TPB, 0, stream>>>(w2, w2h, 128, 576, 576);
  k_cvt_weight<<<nb(256LL * 1152), TPB, 0, stream>>>(w3, w3h, 256, 1152, 1152);
  k_cvt_weight<<<nb(512LL * 2304), TPB, 0, stream>>>(w4, w4h, 512, 2304, 2304);
  k_cvt_weight<<<nb(2048LL * 18432), TPB, 0, stream>>>(fcw1, f1h, 2048, 18432, 18432);
  k_cvt_weight<<<nb(2048LL * 2048), TPB, 0, stream>>>(fcw2, f2h, 2048, 2048, 2048);

  // layer 1: (32,3,128,128) -> conv 64 -> pool (32,64,63,63)   (K=27 padded to 64)
  k_im2col<<<nb(508032LL * 64), TPB, 0, stream>>>(x, imb, 32, 3, 128, 128, 126, 126, 27, 64);
  k_gemm_wmma<<<gemmBlocks(508032, 64), TPB, 0, stream>>>(imb, w1h, gb, 508032, 64, 64);
  k_bias_pool_relu<<<nb(32LL * 64 * 63 * 63), TPB, 0, stream>>>(gb, b1, poolA, 32, 64, 126, 126, 63, 63);

  // layer 2: -> (32,128,30,30)
  k_im2col<<<nb(119072LL * 576), TPB, 0, stream>>>(poolA, imb, 32, 64, 63, 63, 61, 61, 576, 576);
  k_gemm_wmma<<<gemmBlocks(119072, 128), TPB, 0, stream>>>(imb, w2h, gb, 119072, 128, 576);
  k_bias_pool_relu<<<nb(32LL * 128 * 30 * 30), TPB, 0, stream>>>(gb, b2, poolB, 32, 128, 61, 61, 30, 30);

  // layer 3: -> (32,256,14,14)
  k_im2col<<<nb(25088LL * 1152), TPB, 0, stream>>>(poolB, imb, 32, 128, 30, 30, 28, 28, 1152, 1152);
  k_gemm_wmma<<<gemmBlocks(25088, 256), TPB, 0, stream>>>(imb, w3h, gb, 25088, 256, 1152);
  k_bias_pool_relu<<<nb(32LL * 256 * 14 * 14), TPB, 0, stream>>>(gb, b3, poolA, 32, 256, 28, 28, 14, 14);

  // layer 4: -> (32,512,6,6)
  k_im2col<<<nb(4608LL * 2304), TPB, 0, stream>>>(poolA, imb, 32, 256, 14, 14, 12, 12, 2304, 2304);
  k_gemm_wmma<<<gemmBlocks(4608, 512), TPB, 0, stream>>>(imb, w4h, gb, 4608, 512, 2304);
  k_bias_pool_relu<<<nb(32LL * 512 * 6 * 6), TPB, 0, stream>>>(gb, b4, poolB, 32, 512, 12, 12, 6, 6);

  // feat (NCHW flatten == row-major (32,18432)) -> f16
  k_cast_f32_f16<<<nb(589824LL), TPB, 0, stream>>>(poolB, feath, 589824LL);

  // fc1: tanh(feat @ fcw1^T + b)
  k_gemm_wmma<<<gemmBlocks(32, 2048), TPB, 0, stream>>>(feath, f1h, gb, 32, 2048, 18432);
  k_tanh_bias<<<nb(65536LL), TPB, 0, stream>>>(gb, fcb1, z1, z1h, 65536, 2048);

  // fc2: z @ fcw2^T + b
  k_gemm_wmma<<<gemmBlocks(32, 2048), TPB, 0, stream>>>(z1h, f2h, gb, 32, 2048, 2048);
  k_add_bias<<<nb(65536LL), TPB, 0, stream>>>(gb, fcb2, z2, 65536, 2048);

  // heads
  k_heads<<<1, 32, 0, stream>>>(z2, wi, bi, wseg, bs, wrr, brr, initP, segM, ratW, outRatio);

  // trajectory + seg_start_end
  k_traj<<<32, 256, 0, stream>>>(segM, initP, base, outTraj, outSSE);

  // sampling
  k_sample<<<1, 32, 0, stream>>>(ratW, outTraj, maskB, outSampled);
}
